// MP_43173011259759
// MI455X (gfx1250) — matrix operand order repaired and checked by reading, not compiled, not run on hardware
//
#include <hip/hip_runtime.h>
#include <hip/hip_bf16.h>

#define LN_EPS 1e-5f

typedef __bf16 bf16;
typedef __attribute__((ext_vector_type(16))) __bf16 v16bf;
typedef __attribute__((ext_vector_type(8)))  __bf16 v8bf;
typedef __attribute__((ext_vector_type(8)))  float  v8f;

static __device__ __forceinline__ v8f wmma_bf16(v16bf a, v16bf b, v8f c) {
  // D = A(16x32 bf16) x B(32x16 bf16) + C(16x16 f32)
  return __builtin_amdgcn_wmma_f32_16x16x32_bf16(
      /*neg_a=*/false, a, /*neg_b=*/false, b,
      /*c_mod=*/(short)0, c, /*reuse_a=*/false, /*reuse_b=*/false);
}

// A-operand fragment: lane<16 holds K 0-15, lane>=16 holds K 16-31.
// `base` must already include "+ half*16" of the row pointer.
static __device__ __forceinline__ v16bf load_a_frag(const bf16* base) {
  v8bf lo = *(const v8bf*)(base);
  v8bf hi = *(const v8bf*)(base + 8);
  v16bf a;
#pragma unroll
  for (int i = 0; i < 8; i++) { a[i] = lo[i]; a[i + 8] = hi[i]; }
  return a;
}

// ---------------------------------------------------------------- conversions
__global__ void cvt_f32_bf16(const float* __restrict__ in, bf16* __restrict__ out, int n) {
  int i = blockIdx.x * blockDim.x + threadIdx.x;
  if (i < n) out[i] = (bf16)in[i];
}

// in: [z][R][Ccols] f32  ->  out: [z][Ccols][R] bf16   (LDS-tiled transpose)
__global__ void transpose_cvt(const float* __restrict__ in, bf16* __restrict__ out,
                              int R, int Ccols) {
  __shared__ float t[32][33];
  const int c0 = blockIdx.x * 32, r0 = blockIdx.y * 32;
  const float* inp = in + (size_t)blockIdx.z * R * Ccols;
  bf16* outp = out + (size_t)blockIdx.z * R * Ccols;
  const int tx = threadIdx.x, ty = threadIdx.y;  // 32 x 8
#pragma unroll
  for (int j = 0; j < 32; j += 8)
    t[ty + j][tx] = inp[(size_t)(r0 + ty + j) * Ccols + (c0 + tx)];
  __syncthreads();
#pragma unroll
  for (int j = 0; j < 32; j += 8)
    outp[(size_t)(c0 + ty + j) * R + (r0 + tx)] = (bf16)t[tx][ty + j];
}

// ---------------------------------------------------------------- QKV projection
// Register-blocked 32x64 per wave: 2 M-tiles x 4 N-tiles = 8 WMMA per K-step.
// A = xbf [2048,1024]; B from WT (transposed weights, [K,C] per order).
// Writes q,k as bf16 [O,B,H,N,64]; v transposed as [O,B,H,64,N].
__global__ __launch_bounds__(128) void proj_qkv(
    const bf16* __restrict__ xbf,
    const bf16* __restrict__ WqT, const bf16* __restrict__ WkT, const bf16* __restrict__ WvT,
    const float* __restrict__ bq, const float* __restrict__ bk, const float* __restrict__ bv,
    bf16* __restrict__ qo, bf16* __restrict__ ko, bf16* __restrict__ vo) {
  const int o = blockIdx.z / 3, mat = blockIdx.z % 3;
  const int lane = threadIdx.x & 31, wave = threadIdx.x >> 5;
  const int n0 = blockIdx.x * 64;
  const int m0 = (blockIdx.y * 4 + wave) * 32;
  const int half = lane >> 4, r = lane & 15;

  const bf16* WT = (mat == 0 ? WqT : mat == 1 ? WkT : WvT) + (size_t)o * 1024 * 1024;
  const float* bias = (mat == 0 ? bq : mat == 1 ? bk : bv) + o * 1024;

  const bf16* Arow0 = xbf + (size_t)(m0 + r) * 1024 + half * 16;
  const bf16* Arow1 = Arow0 + (size_t)16 * 1024;
  const bf16* Brow[4];
#pragma unroll
  for (int nt = 0; nt < 4; nt++)
    Brow[nt] = WT + (size_t)(n0 + nt * 16 + r) * 1024 + half * 16;

  v8f acc[2][4];
#pragma unroll
  for (int mt = 0; mt < 2; mt++)
#pragma unroll
    for (int nt = 0; nt < 4; nt++) acc[mt][nt] = (v8f){};

  for (int c0 = 0; c0 < 1024; c0 += 32) {
    __builtin_prefetch(Arow0 + c0 + 256, 0, 2);
    __builtin_prefetch(Arow1 + c0 + 256, 0, 2);
    const v16bf a0 = load_a_frag(Arow0 + c0);
    const v16bf a1 = load_a_frag(Arow1 + c0);
#pragma unroll
    for (int nt = 0; nt < 4; nt++) {
      const v16bf bb = *(const v16bf*)(Brow[nt] + c0);
      acc[0][nt] = wmma_bf16(a0, bb, acc[0][nt]);
      acc[1][nt] = wmma_bf16(a1, bb, acc[1][nt]);
    }
  }

#pragma unroll
  for (int nt = 0; nt < 4; nt++) {
    const int kcol = n0 + nt * 16 + r;
    const int h = kcol >> 6, d = kcol & 63;
    const float bsv = bias[kcol];
#pragma unroll
    for (int mt = 0; mt < 2; mt++) {
#pragma unroll
      for (int r2 = 0; r2 < 8; r2++) {
        const int m = m0 + mt * 16 + half * 8 + r2;
        const int b = m >> 10, n = m & 1023;
        float v = acc[mt][nt][r2] + bsv;
        if (o > 0) v = 0.5f * v * (1.0f + erff(v * 0.70710678118f));  // exact GELU
        const bf16 bw = (bf16)v;
        const size_t hb = ((size_t)o * 2 + b) * 16 + h;
        if (mat == 0)      qo[(hb * 1024 + n) * 64 + d] = bw;
        else if (mat == 1) ko[(hb * 1024 + n) * 64 + d] = bw;
        else               vo[(hb * 64 + d) * 1024 + n] = bw;  // V transposed [D,N]
      }
    }
  }
}

// ---------------------------------------------------------------- flash attention
// One 16-query block per wave; stream key blocks of 32; online softmax.
__global__ __launch_bounds__(128) void attn_flash(
    const bf16* __restrict__ qbf, const bf16* __restrict__ kbf, const bf16* __restrict__ vbf,
    const float* __restrict__ order_w, bf16* __restrict__ cat) {
  __shared__ __align__(64) bf16 pt[4][16][32];
  const int ob = blockIdx.z, o = ob >> 1, b = ob & 1;
  const int h = blockIdx.y;
  const int lane = threadIdx.x & 31, wave = threadIdx.x >> 5;
  const int m0 = (blockIdx.x * 4 + wave) * 16;
  const int half = lane >> 4, r = lane & 15;

  const size_t hb = ((size_t)o * 2 + b) * 16 + h;
  const bf16* Q  = qbf + hb * 1024 * 64;   // [N,64]
  const bf16* Kp = kbf + hb * 1024 * 64;   // [N,64]
  const bf16* Vp = vbf + hb * 64 * 1024;   // [64,N]

  v16bf QA[2];
#pragma unroll
  for (int dh = 0; dh < 2; dh++)
    QA[dh] = load_a_frag(Q + (size_t)(m0 + r) * 64 + dh * 32 + half * 16);

  v8f accd[4];
  float mvec[8], lvec[8];
#pragma unroll
  for (int t = 0; t < 4; t++) accd[t] = (v8f){};
#pragma unroll
  for (int i = 0; i < 8; i++) { mvec[i] = -3.0e38f; lvec[i] = 0.f; }

  for (int k0 = 0; k0 < 1024; k0 += 32) {
    v8f Sa = {}, Sb = {};
#pragma unroll
    for (int dh = 0; dh < 2; dh++) {
      const v16bf B0 = *(const v16bf*)(Kp + (size_t)(k0 + r) * 64 + dh * 32 + half * 16);
      Sa = wmma_bf16(QA[dh], B0, Sa);
      const v16bf B1 = *(const v16bf*)(Kp + (size_t)(k0 + 16 + r) * 64 + dh * 32 + half * 16);
      Sb = wmma_bf16(QA[dh], B1, Sb);
    }

    float fvec[8];
#pragma unroll
    for (int r2 = 0; r2 < 8; r2++) {
      const float sa = Sa[r2] * 0.125f, sb = Sb[r2] * 0.125f;  // D^-0.5, D=64
      float mx = fmaxf(sa, sb);
#pragma unroll
      for (int msk = 1; msk < 16; msk <<= 1) mx = fmaxf(mx, __shfl_xor(mx, msk, 32));
      const float mnew = fmaxf(mvec[r2], mx);
      const float pa = __expf(sa - mnew), pb = __expf(sb - mnew);
      float rs = pa + pb;
#pragma unroll
      for (int msk = 1; msk < 16; msk <<= 1) rs += __shfl_xor(rs, msk, 32);
      const float f = __expf(mvec[r2] - mnew);
      lvec[r2] = lvec[r2] * f + rs;
      mvec[r2] = mnew;
      fvec[r2] = f;
      pt[wave][half * 8 + r2][r]      = (bf16)pa;   // C-layout -> LDS [row][col]
      pt[wave][half * 8 + r2][16 + r] = (bf16)pb;
    }
#pragma unroll
    for (int t = 0; t < 4; t++)
#pragma unroll
      for (int r2 = 0; r2 < 8; r2++) accd[t][r2] *= fvec[r2];

    // Re-read P in A-operand layout (same wave; LDS ops in order)
    const v16bf PA = load_a_frag(&pt[wave][r][half * 16]);
#pragma unroll
    for (int t = 0; t < 4; t++) {
      const v16bf VB = *(const v16bf*)(Vp + (size_t)(t * 16 + r) * 1024 + k0 + half * 16);
      accd[t] = wmma_bf16(PA, VB, accd[t]);
    }
  }

  const float ow = order_w[o];
#pragma unroll
  for (int r2 = 0; r2 < 8; r2++) {
    const int n = m0 + half * 8 + r2;
    const float inv = ow / lvec[r2];
#pragma unroll
    for (int t = 0; t < 4; t++) {
      const int d = t * 16 + r;
      cat[((size_t)b * 1024 + n) * 3072 + o * 1024 + h * 64 + d] = (bf16)(accd[t][r2] * inv);
    }
  }
}

// ---------------------------------------------------------------- output projection
// Register-blocked 32x64 per wave, K = 3072.
__global__ __launch_bounds__(128) void out_proj(
    const bf16* __restrict__ cat, const bf16* __restrict__ WoT,
    const float* __restrict__ bo, float* __restrict__ y) {
  const int lane = threadIdx.x & 31, wave = threadIdx.x >> 5;
  const int n0 = blockIdx.x * 64;
  const int m0 = (blockIdx.y * 4 + wave) * 32;
  const int half = lane >> 4, r = lane & 15;

  const bf16* Arow0 = cat + (size_t)(m0 + r) * 3072 + half * 16;
  const bf16* Arow1 = Arow0 + (size_t)16 * 3072;
  const bf16* Brow[4];
#pragma unroll
  for (int nt = 0; nt < 4; nt++)
    Brow[nt] = WoT + (size_t)(n0 + nt * 16 + r) * 3072 + half * 16;

  v8f acc[2][4];
#pragma unroll
  for (int mt = 0; mt < 2; mt++)
#pragma unroll
    for (int nt = 0; nt < 4; nt++) acc[mt][nt] = (v8f){};

  for (int c0 = 0; c0 < 3072; c0 += 32) {
    __builtin_prefetch(Arow0 + c0 + 256, 0, 2);
    __builtin_prefetch(Arow1 + c0 + 256, 0, 2);
    const v16bf a0 = load_a_frag(Arow0 + c0);
    const v16bf a1 = load_a_frag(Arow1 + c0);
#pragma unroll
    for (int nt = 0; nt < 4; nt++) {
      const v16bf bb = *(const v16bf*)(Brow[nt] + c0);
      acc[0][nt] = wmma_bf16(a0, bb, acc[0][nt]);
      acc[1][nt] = wmma_bf16(a1, bb, acc[1][nt]);
    }
  }

#pragma unroll
  for (int nt = 0; nt < 4; nt++) {
    const int col = n0 + nt * 16 + r;
    const float bias = bo[col];
#pragma unroll
    for (int mt = 0; mt < 2; mt++) {
#pragma unroll
      for (int r2 = 0; r2 < 8; r2++) {
        const int m = m0 + mt * 16 + half * 8 + r2;
        y[(size_t)m * 1024 + col] = acc[mt][nt][r2] + bias;
      }
    }
  }
}

// ---------------------------------------------------------------- layernorm
__global__ __launch_bounds__(256) void layernorm_kernel(
    const float* __restrict__ y, const float* __restrict__ gamma,
    const float* __restrict__ beta, float* __restrict__ out) {
  __shared__ float ssum[8], ssq[8];
  const int row = blockIdx.x, tid = threadIdx.x;
  const float* yr = y + (size_t)row * 1024;
  float s = 0.f, q = 0.f;
  for (int c = tid; c < 1024; c += 256) { const float v = yr[c]; s += v; q += v * v; }
#pragma unroll
  for (int msk = 1; msk < 32; msk <<= 1) { s += __shfl_xor(s, msk, 32); q += __shfl_xor(q, msk, 32); }
  if ((tid & 31) == 0) { ssum[tid >> 5] = s; ssq[tid >> 5] = q; }
  __syncthreads();
  s = 0.f; q = 0.f;
#pragma unroll
  for (int i = 0; i < 8; i++) { s += ssum[i]; q += ssq[i]; }
  const float mu = s * (1.0f / 1024.0f);
  const float var = q * (1.0f / 1024.0f) - mu * mu;
  const float rstd = rsqrtf(var + LN_EPS);
  for (int c = tid; c < 1024; c += 256)
    out[(size_t)row * 1024 + c] = (yr[c] - mu) * rstd * gamma[c] + beta[c];
}

// ---------------------------------------------------------------- launch
extern "C" void kernel_launch(void* const* d_in, const int* in_sizes, int n_in,
                              void* d_out, int out_size, void* d_ws, size_t ws_size,
                              hipStream_t stream) {
  const float* x     = (const float*)d_in[0];
  const float* Wq    = (const float*)d_in[1];
  const float* bq    = (const float*)d_in[2];
  const float* Wk    = (const float*)d_in[3];
  const float* bk    = (const float*)d_in[4];
  const float* Wv    = (const float*)d_in[5];
  const float* bv    = (const float*)d_in[6];
  const float* ow    = (const float*)d_in[7];
  const float* Wo    = (const float*)d_in[8];
  const float* bo    = (const float*)d_in[9];
  const float* gamma = (const float*)d_in[10];
  const float* beta  = (const float*)d_in[11];
  float* out = (float*)d_out;

  char* ws = (char*)d_ws;
  size_t off = 0;
  auto alloc = [&](size_t bytes) -> char* {
    char* p = ws + off;
    off += (bytes + 255) & ~(size_t)255;
    return p;
  };
  bf16* xbf  = (bf16*)alloc((size_t)2048 * 1024 * 2);
  bf16* WqT  = (bf16*)alloc((size_t)3 * 1024 * 1024 * 2);
  bf16* WkT  = (bf16*)alloc((size_t)3 * 1024 * 1024 * 2);
  bf16* WvT  = (bf16*)alloc((size_t)3 * 1024 * 1024 * 2);
  bf16* WoT  = (bf16*)alloc((size_t)1024 * 3072 * 2);
  bf16* qb_  = (bf16*)alloc((size_t)3 * 2 * 16 * 1024 * 64 * 2);
  bf16* kb_  = (bf16*)alloc((size_t)3 * 2 * 16 * 1024 * 64 * 2);
  bf16* vb_  = (bf16*)alloc((size_t)3 * 2 * 16 * 64 * 1024 * 2);
  bf16* cat  = (bf16*)alloc((size_t)2048 * 3072 * 2);
  float* ybf = (float*)alloc((size_t)2048 * 1024 * 4);

  const dim3 tb(32, 8);
  cvt_f32_bf16<<<(2048 * 1024) / 256, 256, 0, stream>>>(x, xbf, 2048 * 1024);
  transpose_cvt<<<dim3(32, 32, 3), tb, 0, stream>>>(Wq, WqT, 1024, 1024);
  transpose_cvt<<<dim3(32, 32, 3), tb, 0, stream>>>(Wk, WkT, 1024, 1024);
  transpose_cvt<<<dim3(32, 32, 3), tb, 0, stream>>>(Wv, WvT, 1024, 1024);
  transpose_cvt<<<dim3(32, 96, 1), tb, 0, stream>>>(Wo, WoT, 3072, 1024);

  proj_qkv<<<dim3(16, 16, 9), 128, 0, stream>>>(xbf, WqT, WkT, WvT, bq, bk, bv, qb_, kb_, vb_);
  attn_flash<<<dim3(16, 16, 6), 128, 0, stream>>>(qb_, kb_, vb_, ow, cat);
  out_proj<<<dim3(16, 16, 1), 128, 0, stream>>>(cat, WoT, bo, ybf);
  layernorm_kernel<<<2048, 256, 0, stream>>>(ybf, gamma, beta, out);
}
